// qNetwork_SNN_39771397161539
// MI455X (gfx1250) — compile-verified, matrix operand order checked by reading, not run on hardware
//
#include <hip/hip_runtime.h>
#include <stdint.h>

typedef __attribute__((ext_vector_type(16))) _Float16 v16h;
typedef __attribute__((ext_vector_type(8)))  float    v8f;

#define BETA 0.95f
#define THR  1.0f
#define TSTEPS 25

static __device__ __forceinline__ v8f wmma16(v16h a, v16h b, v8f c) {
  // D = A(16x32 f16) * B(32x16 f16) + C(16x16 f32)
  return __builtin_amdgcn_wmma_f32_16x16x32_f16(false, a, false, b, (short)0, c,
                                                false, false);
}

// k-offset within a 32-wide A/B fragment tile for (vgpr-pair v, elem e, lane-group g)
// per CDNA5 ISA 7.12.2 (16-bit A 16x32 layout).
static __device__ __forceinline__ int kk_of(int v, int e, int g) {
  return (v < 4) ? (8 * g + 2 * v + e) : (16 + 8 * g + 2 * (v - 4) + e);
}

// ---------------------------------------------------------------------------
// Pack a row-major f32 matrix [R x K] into f16 WMMA fragments.
// Fragment element index: ((rt*KT + kt)*32 + lane)*16 + (v*2 + e)
// holds src[r=rt*16+(lane&15)][k=kt*32+kk(v,e,lane>>4)].
// Same math packs A-matrices (rows = batch) and B-matrices (rows = out-feat,
// since cur = in @ W.T means B[k][n] = W[n][k]).
// ---------------------------------------------------------------------------
__global__ void pack_frag(const float* __restrict__ src, _Float16* __restrict__ dst,
                          int K) {
  int idx    = blockIdx.x * 256 + threadIdx.x;
  int within = idx & 15;
  int lane   = (idx >> 4) & 31;
  int tile   = idx >> 9;
  int KT     = K >> 5;
  int kt     = tile % KT;
  int rt     = tile / KT;
  int e      = within & 1;
  int v      = within >> 1;
  int g      = lane >> 4;
  int r      = rt * 16 + (lane & 15);
  int k      = kt * 32 + kk_of(v, e, g);
  dst[idx] = (_Float16)src[(size_t)r * K + k];
}

// ---------------------------------------------------------------------------
// Layer-0 GEMM (loop-invariant): cur0 = x @ W0.T + b0, computed once.
// Output stored in the per-thread fragment order the main kernel's phase-A
// threads consume: cur0p[(mtile*256 + row*16 + q)*64 + i],
//   thread (row,q) covers k in [64q,64q+64), i = c2*32 + gk*16 + v*2 + e.
// ---------------------------------------------------------------------------
__global__ void __launch_bounds__(256) l0_cur(const _Float16* __restrict__ Xp,
                                              const _Float16* __restrict__ W0p,
                                              const float* __restrict__ b0,
                                              float* __restrict__ cur0p) {
  int t = threadIdx.x;
  int w = t >> 5, l = t & 31, col = l & 15, g = l >> 4;
  int nt = blockIdx.x * 8 + w;   // 0..63
  int mt = blockIdx.y;           // 0..511

  v8f acc = {0.f, 0.f, 0.f, 0.f, 0.f, 0.f, 0.f, 0.f};
  for (int kt = 0; kt < 16; ++kt) {  // K = 512
    v16h a = *(const v16h*)(Xp  + ((size_t)mt * 16 + kt) * 512 + l * 16);
    v16h b = *(const v16h*)(W0p + ((size_t)nt * 16 + kt) * 512 + l * 16);
    acc = wmma16(a, b, acc);
  }

  int n    = nt * 16 + col;
  float bi = b0[n];
  // destination decomposition for column n
  int q    = n >> 6;
  int c2   = (n >> 5) & 1;
  int kmod = n & 31;
  int e    = kmod & 1;
  int gk   = (kmod >> 3) & 1;
  int hi   = kmod >> 4;
  int v2   = (hi ? 4 : 0) + ((kmod & 7) >> 1);
  int i    = c2 * 32 + gk * 16 + v2 * 2 + e;
#pragma unroll
  for (int r = 0; r < 8; ++r) {
    int row = r + 8 * g;
    cur0p[((size_t)mt * 256 + row * 16 + q) * 64 + i] = acc[r] + bi;
  }
}

// ---------------------------------------------------------------------------
// Main persistent SNN kernel: one 8-wave workgroup owns 16 batch rows for all
// 25 timesteps. Membranes live in VGPRs; spikes go through LDS pre-swizzled
// as WMMA A-fragments (2 x 32 KB).
// ---------------------------------------------------------------------------
__global__ void __launch_bounds__(256, 1) snn_main(const float* __restrict__ cur0p,
                                                   const _Float16* __restrict__ W1p,
                                                   const float* __restrict__ b1,
                                                   const _Float16* __restrict__ W2p,
                                                   const float* __restrict__ b2,
                                                   float* __restrict__ out) {
  __shared__ __align__(32) uint16_t spk0h[16384];  // 32 KB: 32 k-tiles of frags
  __shared__ __align__(32) uint16_t spk1h[16384];  // 32 KB
  unsigned int* spk0w = (unsigned int*)spk0h;

  const int t     = threadIdx.x;
  const int mtile = blockIdx.x;
  // phase-A identity: row m, k-range q
  const int m = t >> 4, q = t & 15;
  // wave identity for GEMM phases
  const int w = t >> 5, l = t & 31, col = l & 15, g = l >> 4;

  float mem0r[64];                 // layer-0 membrane, frag order
  float mem1r[64];                 // layer-1 membrane: [j(ntile)][r(row)]
  float mem2r[16], accr[16];       // layer-2 membrane + spike accumulator
#pragma unroll
  for (int i = 0; i < 64; ++i) { mem0r[i] = 0.f; mem1r[i] = 0.f; }
#pragma unroll
  for (int i = 0; i < 16; ++i) { mem2r[i] = 0.f; accr[i] = 0.f; }

  float bias1r[8], bias2r[2];
#pragma unroll
  for (int j = 0; j < 8; ++j) bias1r[j] = b1[(w * 8 + j) * 16 + col];
#pragma unroll
  for (int j = 0; j < 2; ++j) bias2r[j] = b2[(2 * w + j) * 16 + col];

  const float* curbase = cur0p + ((size_t)mtile * 256 + t) * 64;

  // phase-B spike-store decomposition of column n = (w*8+j)*16+col:
  const int e1  = col & 1;
  const int gk1 = (col >> 3) & 1;
  const int vlo = (col & 7) >> 1;

  const v8f zero8 = {0.f, 0.f, 0.f, 0.f, 0.f, 0.f, 0.f, 0.f};

  for (int step = 0; step < TSTEPS; ++step) {
    // ---- Phase A: layer-0 LIF update (cur0 is precomputed), emit spk0 frags
#pragma unroll
    for (int c2 = 0; c2 < 2; ++c2) {
      int kt = 2 * q + c2;
#pragma unroll
      for (int gg = 0; gg < 2; ++gg) {
#pragma unroll
        for (int v = 0; v < 8; ++v) {
          int i0 = c2 * 32 + gg * 16 + v * 2;
          unsigned int pk = 0u;
#pragma unroll
          for (int e = 0; e < 2; ++e) {
            float cur   = curbase[i0 + e];
            float mm    = mem0r[i0 + e];
            float reset = (mm > THR) ? THR : 0.f;
            mm = BETA * mm + cur - reset;
            mem0r[i0 + e] = mm;
            if (mm > THR) pk |= (0x3C00u << (16 * e));  // f16 1.0
          }
          spk0w[kt * 256 + (m + 16 * gg) * 8 + v] = pk;
        }
      }
    }
    __syncthreads();

    // ---- Phase B: layer-1 GEMM (8 n-tiles per wave) + in-lane LIF update
    v8f acc[8];
#pragma unroll
    for (int j = 0; j < 8; ++j) acc[j] = zero8;
    for (int kt = 0; kt < 32; ++kt) {  // K = 1024
      v16h a = *(const v16h*)(&spk0h[kt * 512 + l * 16]);
#pragma unroll
      for (int j = 0; j < 8; ++j) {
        v16h b = *(const v16h*)(W1p + ((size_t)(w * 8 + j) * 32 + kt) * 512 + l * 16);
        acc[j] = wmma16(a, b, acc[j]);
      }
    }
#pragma unroll
    for (int j = 0; j < 8; ++j) {
      int hi  = j & 1;
      int v2  = vlo + (hi ? 4 : 0);
      int kt2 = w * 4 + (j >> 1);
#pragma unroll
      for (int r = 0; r < 8; ++r) {
        float cur   = acc[j][r] + bias1r[j];
        float mm    = mem1r[j * 8 + r];
        float reset = (mm > THR) ? THR : 0.f;
        mm = BETA * mm + cur - reset;
        mem1r[j * 8 + r] = mm;
        uint16_t s = (mm > THR) ? (uint16_t)0x3C00 : (uint16_t)0;
        int row = r + 8 * g;
        spk1h[kt2 * 512 + (row + 16 * gk1) * 16 + v2 * 2 + e1] = s;
      }
    }
    __syncthreads();

    // ---- Phase C: layer-2 GEMM (2 n-tiles per wave) + LIF + accumulate
    v8f acc2[2];
#pragma unroll
    for (int j = 0; j < 2; ++j) acc2[j] = zero8;
    for (int kt = 0; kt < 32; ++kt) {  // K = 1024
      v16h a = *(const v16h*)(&spk1h[kt * 512 + l * 16]);
#pragma unroll
      for (int j = 0; j < 2; ++j) {
        v16h b = *(const v16h*)(W2p + ((size_t)(2 * w + j) * 32 + kt) * 512 + l * 16);
        acc2[j] = wmma16(a, b, acc2[j]);
      }
    }
#pragma unroll
    for (int j = 0; j < 2; ++j) {
#pragma unroll
      for (int r = 0; r < 8; ++r) {
        float cur   = acc2[j][r] + bias2r[j];
        float mm    = mem2r[j * 8 + r];
        float reset = (mm > THR) ? THR : 0.f;
        mm = BETA * mm + cur - reset;
        mem2r[j * 8 + r] = mm;
        accr[j * 8 + r] += (mm > THR) ? 1.f : 0.f;
      }
    }
    __syncthreads();
  }

  // ---- Write accumulated last-layer spike counts
#pragma unroll
  for (int j = 0; j < 2; ++j) {
    int n = (2 * w + j) * 16 + col;
#pragma unroll
    for (int r = 0; r < 8; ++r) {
      out[((size_t)mtile * 16 + r + 8 * g) * 256 + n] = accr[j * 8 + r];
    }
  }
}

// ---------------------------------------------------------------------------
extern "C" void kernel_launch(void* const* d_in, const int* in_sizes, int n_in,
                              void* d_out, int out_size, void* d_ws, size_t ws_size,
                              hipStream_t stream) {
  (void)in_sizes; (void)n_in; (void)out_size; (void)ws_size;
  const float* x  = (const float*)d_in[0];
  const float* W0 = (const float*)d_in[1];
  const float* b0 = (const float*)d_in[2];
  const float* W1 = (const float*)d_in[3];
  const float* b1 = (const float*)d_in[4];
  const float* W2 = (const float*)d_in[5];
  const float* b2 = (const float*)d_in[6];

  // Workspace layout (~43.5 MB):
  _Float16* Xp  = (_Float16*)d_ws;                  // 8192*512  f16 (8 MB)
  _Float16* W0p = Xp  + (size_t)8192 * 512;         // 1024*512  f16 (1 MB)
  _Float16* W1p = W0p + (size_t)1024 * 512;         // 1024*1024 f16 (2 MB)
  _Float16* W2p = W1p + (size_t)1024 * 1024;        //  256*1024 f16 (0.5 MB)
  float*  cur0p = (float*)(W2p + (size_t)256 * 1024); // 8192*1024 f32 (32 MB)

  pack_frag<<<(8192 * 512) / 256, 256, 0, stream>>>(x,  Xp,  512);
  pack_frag<<<(1024 * 512) / 256, 256, 0, stream>>>(W0, W0p, 512);
  pack_frag<<<(1024 * 1024) / 256, 256, 0, stream>>>(W1, W1p, 1024);
  pack_frag<<<(256 * 1024) / 256, 256, 0, stream>>>(W2, W2p, 1024);

  l0_cur<<<dim3(8, 512), 256, 0, stream>>>(Xp, W0p, b0, cur0p);

  snn_main<<<512, 256, 0, stream>>>(cur0p, W1p, b1, W2p, b2, (float*)d_out);
}
